// GruKerasModel_88734024335639
// MI455X (gfx1250) — compile-verified, matrix operand order checked by reading, not run on hardware
//
#include <hip/hip_runtime.h>
#include <hip/hip_bf16.h>

// ---------------- problem constants ----------------
#define B_  256
#define T_  15
#define L_  6
#define U_  1000
#define E_  1024
#define V_  32000
#define TU_ 3000   // 3*U

// ---------------- vector types ----------------
typedef __attribute__((ext_vector_type(16))) __bf16 v16bf;
typedef __attribute__((ext_vector_type(8)))  __bf16 v8bf;
typedef __attribute__((ext_vector_type(8)))  float  v8f;
typedef __attribute__((ext_vector_type(4)))  float  v4f;
typedef __attribute__((ext_vector_type(4)))  unsigned int v4u;
typedef __attribute__((ext_vector_type(8)))  int    v8i;
typedef __attribute__((ext_vector_type(4)))  int    v4i;

__device__ __forceinline__ __bf16 to_bf16(float f)  { return (__bf16)f; }
__device__ __forceinline__ __bf16 to_bf16(__bf16 f) { return f; }

// ---------------- conversion / gather kernels ----------------
__global__ void k_f32_to_bf16(const float* __restrict__ src, __bf16* __restrict__ dst, long n) {
  long i = (long)blockIdx.x * blockDim.x + threadIdx.x;
  long stride = (long)gridDim.x * blockDim.x;
  for (; i < n; i += stride) dst[i] = to_bf16(src[i]);
}

// X[t*B + b][e] = bf16(emb[tokens[b*T + t]][e])
__global__ void k_embed(const int* __restrict__ tokens, const float* __restrict__ emb,
                        __bf16* __restrict__ X) {
  int row = blockIdx.x;              // row = t*B + b
  int t = row / B_, b = row % B_;
  int tok = tokens[b * T_ + t];
  const float* src = emb + (size_t)tok * E_;
  __bf16* dst = X + (size_t)row * E_;
  for (int e = threadIdx.x; e < E_; e += blockDim.x) dst[e] = to_bf16(src[e]);
}

// init_states [L][B][U] -> Hf32 (same layout) and Hcat bf16 [B][L*U]
__global__ void k_init_h(const float* __restrict__ init, float* __restrict__ Hf,
                         __bf16* __restrict__ Hcat) {
  long i = (long)blockIdx.x * blockDim.x + threadIdx.x;
  if (i >= (long)L_ * B_ * U_) return;
  int l = (int)(i / ((long)B_ * U_));
  int rem = (int)(i % ((long)B_ * U_));
  int b = rem / U_, u = rem % U_;
  float v = init[i];
  Hf[i] = v;
  Hcat[(size_t)b * (L_ * U_) + l * U_ + u] = to_bf16(v);
}

// ---------------- bf16 WMMA GEMM ----------------
// C[M=256, N] (f32) = A[256, K] (bf16, lda) x B[K, N] (TB, ldb)  [+ bias] [+ C]
// flags: bit0 = add bias[col], bit1 = accumulate into existing C
#define BM 128
#define BN 128
#define BK 32
#define LDS_S 40   // padded LDS row stride (elements): 64B data + 16B pad per row

// store 8 consecutive-n B elements into the transposed LDS tile
__device__ __forceinline__ void stage8(__bf16* Bs, int n0, int c, const __bf16* src) {
  v8bf v = *(const v8bf*)src;          // one b128 global load
#pragma unroll
  for (int e = 0; e < 8; ++e) Bs[(n0 + e) * LDS_S + c] = v[e];
}
__device__ __forceinline__ void stage8(__bf16* Bs, int n0, int c, const float* src) {
  v4f a = *(const v4f*)src;            // two b128 global loads + cvt
  v4f b = *(const v4f*)(src + 4);
#pragma unroll
  for (int e = 0; e < 4; ++e) {
    Bs[(n0 + e)     * LDS_S + c] = to_bf16(a[e]);
    Bs[(n0 + e + 4) * LDS_S + c] = to_bf16(b[e]);
  }
}

template <typename TB>
__global__ __launch_bounds__(256)
void k_gemm(const __bf16* __restrict__ A, int lda,
            const TB* __restrict__ Bsrc, int ldb,
            float* __restrict__ C, int ldc,
            const float* __restrict__ bias,
            int N, int K, int flags) {
  __shared__ __align__(16) __bf16 As[BM * LDS_S];   // [BM][BK] row-major (padded by TDM)
  __shared__ __align__(16) __bf16 Bs[BN * LDS_S];   // B transposed: [BN][BK] (padded)

  const int tid  = threadIdx.x;
  const int lane = tid & 31;
  const int wave = tid >> 5;          // 8 wave32s
  const int wm   = wave >> 1;         // 4 wave-rows (32 rows each)
  const int wn   = wave & 1;          // 2 wave-cols (64 cols each)
  const int half = lane >> 4;         // ISA 16-bit A/B layout: lanes 0-15 vs 16-31
  const int l16  = lane & 15;

  const int rowTile = blockIdx.y * BM;   // M=256 is an exact multiple of BM
  const int colTile = blockIdx.x * BN;
  const bool colFull = (colTile + BN <= N);

  // per-wave TDM slice of the A tile: 16 rows of 32 elements
  const unsigned ldsA = (unsigned)(uintptr_t)(&As[0]) + (unsigned)(wave * 16 * LDS_S * 2);
  const __bf16* Arow = A + (size_t)(rowTile + wave * 16) * lda;

  v8f acc[2][4] = {};

  for (int k0 = 0; k0 < K; k0 += BK) {
    // ---- A tile via Tensor Data Mover: 16x32 bf16 sub-tile per wave ----
    // tensor_dim0 = K-k0  -> hardware zero-fills reads past the K tail
    // pad: after every 16 DWORDs (one 64B row) insert 4 DWORDs (16B) -> LDS_S=40 stride
    {
      unsigned long long ga = (unsigned long long)(uintptr_t)(Arow + k0);
      unsigned tdim0 = (unsigned)(K - k0);
      v4u g0 = { 1u,                                         // count=1 (valid descriptor)
                 ldsA,                                       // lds_addr
                 (unsigned)(ga & 0xFFFFFFFFu),               // global_addr[31:0]
                 (unsigned)((ga >> 32) & 0x01FFFFFFu) | (2u << 30) }; // addr[56:32], type=2
      v8i g1 = { (int)((1u << 16)      // data_size = 2 bytes
                     | (1u << 20)      // pad_enable
                     | (3u << 22)      // pad_interval = 16 DWORDs
                     | (3u << 25)),    // pad_amount  = 4 DWORDs
                 (int)((tdim0 & 0xFFFFu) << 16),                    // tensor_dim0[15:0]
                 (int)(((tdim0 >> 16) & 0xFFFFu) | (16u << 16)),    // dim0[31:16], tensor_dim1=16
                 (int)(BK << 16),                                   // tile_dim0 = 32
                 16,                                                // tile_dim1 = 16
                 lda,                                               // tensor_dim0_stride
                 0, 0 };
      v4i gz  = { 0, 0, 0, 0 };
      v8i gz8 = { 0, 0, 0, 0, 0, 0, 0, 0 };
      __builtin_amdgcn_tensor_load_to_lds(g0, g1, gz, gz, gz8, 0);
    }

    // ---- B tile into LDS transposed ----
    if (colFull && (k0 + BK <= K)) {
      // fast path: vectorized, no bounds checks (covers ~31/32 of all iterations)
#pragma unroll
      for (int it = 0; it < 2; ++it) {
        int idx = tid + it * 256;              // 512 chunks of 8 elements
        int c = idx >> 4, n0 = (idx & 15) << 3;
        stage8(Bs, n0, c, Bsrc + (size_t)(k0 + c) * ldb + colTile + n0);
      }
    } else {
      // guarded path: K tail and/or partial N tile, zero-filled
      for (int idx = tid; idx < BN * BK; idx += 256) {
        int c = idx / BN, n = idx % BN;
        int gk = k0 + c, gn = colTile + n;
        Bs[n * LDS_S + c] =
            (gk < K && gn < N) ? to_bf16(Bsrc[(size_t)gk * ldb + gn]) : to_bf16(0.0f);
      }
    }
    // speculative prefetch of the next K-slab of B (global_prefetch_b8)
    if (k0 + BK < K)
      __builtin_prefetch(&Bsrc[(size_t)(k0 + BK) * ldb + colTile + (tid & 127)], 0, 1);

    __builtin_amdgcn_s_wait_tensorcnt(0);   // this wave's A sub-tile is in LDS
    __syncthreads();                        // all waves' A + B tiles visible

    // ---- fragment loads (two ds_load_b128 each) + 8 WMMAs per wave ----
    v16bf a[2], b[4];
#pragma unroll
    for (int mi = 0; mi < 2; ++mi) {
      const __bf16* p = As + (wm * 32 + mi * 16 + l16) * LDS_S;
      union { v16bf v; v8bf h[2]; } u;
      u.h[0] = *(const v8bf*)(p + half * 8);        // K 0-7   (lanes<16) / 8-15
      u.h[1] = *(const v8bf*)(p + 16 + half * 8);   // K 16-23 (lanes<16) / 24-31
      a[mi] = u.v;
    }
#pragma unroll
    for (int ni = 0; ni < 4; ++ni) {
      const __bf16* p = Bs + (wn * 64 + ni * 16 + l16) * LDS_S;
      union { v16bf v; v8bf h[2]; } u;
      u.h[0] = *(const v8bf*)(p + half * 8);
      u.h[1] = *(const v8bf*)(p + 16 + half * 8);
      b[ni] = u.v;
    }
#pragma unroll
    for (int mi = 0; mi < 2; ++mi)
#pragma unroll
      for (int ni = 0; ni < 4; ++ni)
        acc[mi][ni] = __builtin_amdgcn_wmma_f32_16x16x32_bf16(
            false, a[mi], false, b[ni], (short)0, acc[mi][ni], false, false);
    __syncthreads();
  }

  // ---- epilogue: C/D layout -> VGPR v holds (M = v + half*8, N = l16) per 16x16 tile ----
#pragma unroll
  for (int mi = 0; mi < 2; ++mi) {
#pragma unroll
    for (int ni = 0; ni < 4; ++ni) {
      int col = colTile + wn * 64 + ni * 16 + l16;
      if (col < N) {
#pragma unroll
        for (int v = 0; v < 8; ++v) {
          int row = rowTile + wm * 32 + mi * 16 + half * 8 + v;
          float val = acc[mi][ni][v];
          if (flags & 1) val += bias[col];
          float* cp = C + (size_t)row * ldc + col;
          if (flags & 2) val += *cp;
          *cp = val;
        }
      }
    }
  }
}

// ---------------- GRU gate math (Keras reset_after=True, gate order z,r,h) ----------------
__global__ void k_gru_elem(const float* __restrict__ mx, const float* __restrict__ mh,
                           float* __restrict__ hf, __bf16* __restrict__ hcat, int ldh) {
  int i = blockIdx.x * blockDim.x + threadIdx.x;
  if (i >= B_ * U_) return;
  int b = i / U_, j = i % U_;
  const float* mxr = mx + (size_t)b * TU_;
  const float* mhr = mh + (size_t)b * TU_;
  float z    = 1.0f / (1.0f + __expf(-(mxr[j]           + mhr[j])));
  float r    = 1.0f / (1.0f + __expf(-(mxr[U_ + j]      + mhr[U_ + j])));
  float cand = tanhf(mxr[2 * U_ + j] + r * mhr[2 * U_ + j]);
  float h  = hf[i];
  float hn = z * h + (1.0f - z) * cand;
  hf[i] = hn;                                   // f32 master copy
  hcat[(size_t)b * ldh + j] = to_bf16(hn);      // bf16 copy inside Hcat (next GEMM's A)
}

// ---------------- row softmax (in-place), one block per row ----------------
__global__ void k_softmax(float* __restrict__ x, int N) {
  __shared__ float red[256];
  float* p = x + (size_t)blockIdx.x * N;
  int tid = threadIdx.x;

  float m = -3.402823466e+38f;
  for (int j = tid; j < N; j += 256) m = fmaxf(m, p[j]);
  red[tid] = m; __syncthreads();
  for (int s = 128; s > 0; s >>= 1) {
    if (tid < s) red[tid] = fmaxf(red[tid], red[tid + s]);
    __syncthreads();
  }
  m = red[0]; __syncthreads();

  float sum = 0.0f;
  for (int j = tid; j < N; j += 256) sum += __expf(p[j] - m);
  red[tid] = sum; __syncthreads();
  for (int s = 128; s > 0; s >>= 1) {
    if (tid < s) red[tid] += red[tid + s];
    __syncthreads();
  }
  float inv = 1.0f / red[0];
  for (int j = tid; j < N; j += 256) p[j] = __expf(p[j] - m) * inv;
}

// ---------------- host orchestration ----------------
extern "C" void kernel_launch(void* const* d_in, const int* in_sizes, int n_in,
                              void* d_out, int out_size, void* d_ws, size_t ws_size,
                              hipStream_t stream) {
  (void)in_sizes; (void)n_in; (void)out_size; (void)ws_size;

  const int*   tokens   = (const int*)d_in[0];    // [B,T]
  const float* init_st  = (const float*)d_in[1];  // [L,B,U]
  const float* emb      = (const float*)d_in[2];  // [V,E]
  const float* kernel0  = (const float*)d_in[3];  // [E,3U]
  const float* krest    = (const float*)d_in[4];  // [L-1, E+U, 3U]
  const float* reck     = (const float*)d_in[5];  // [L, U, 3U]
  const float* bias_in  = (const float*)d_in[6];  // [L,3U]
  const float* bias_rec = (const float*)d_in[7];  // [L,3U]
  const float* dense_w  = (const float*)d_in[8];  // [L*U, V]
  const float* dense_b  = (const float*)d_in[9];  // [V]
  float* out = (float*)d_out;                     // [B,V] softmax

  // workspace carve-up (256B aligned); total ~126 MB
  char* ws = (char*)d_ws;
  size_t off = 0;
  auto take = [&](size_t bytes) -> char* {
    off = (off + 255) & ~(size_t)255;
    char* p = ws + off;
    off += bytes;
    return p;
  };
  __bf16* k0_bf    = (__bf16*)take((size_t)E_ * TU_ * 2);
  __bf16* krest_bf = (__bf16*)take((size_t)(L_ - 1) * (E_ + U_) * TU_ * 2);
  __bf16* rec_bf   = (__bf16*)take((size_t)L_ * U_ * TU_ * 2);
  __bf16* X        = (__bf16*)take((size_t)T_ * B_ * E_ * 2);
  __bf16* Hcat     = (__bf16*)take((size_t)B_ * L_ * U_ * 2);
  float*  Hf32     = (float*) take((size_t)L_ * B_ * U_ * 4);
  float*  mx       = (float*) take((size_t)B_ * TU_ * 4);
  float*  mh       = (float*) take((size_t)B_ * TU_ * 4);

  // one-time bf16 conversion of the 15x-reused GRU weights (103 MB -> resident in 192MB L2)
  k_f32_to_bf16<<<2048, 256, 0, stream>>>(kernel0, k0_bf, (long)E_ * TU_);
  k_f32_to_bf16<<<4096, 256, 0, stream>>>(krest, krest_bf, (long)(L_ - 1) * (E_ + U_) * TU_);
  k_f32_to_bf16<<<4096, 256, 0, stream>>>(reck, rec_bf, (long)L_ * U_ * TU_);
  k_embed<<<T_ * B_, 256, 0, stream>>>(tokens, emb, X);
  {
    long n = (long)L_ * B_ * U_;
    k_init_h<<<(unsigned)((n + 255) / 256), 256, 0, stream>>>(init_st, Hf32, Hcat);
  }

  dim3 blk(256);
  dim3 grid_gru((TU_ + BN - 1) / BN, B_ / BM);   // 24 x 2 blocks

  for (int t = 0; t < T_; ++t) {
    const __bf16* xt = X + (size_t)t * B_ * E_;
    for (int l = 0; l < L_; ++l) {
      const float* bi = bias_in + (size_t)l * TU_;
      const float* br = bias_rec + (size_t)l * TU_;
      // mx = x_t @ Kx + bi  (+ prev @ Kh for l>=1, accumulated in a second GEMM)
      if (l == 0) {
        k_gemm<__bf16><<<grid_gru, blk, 0, stream>>>(
            xt, E_, k0_bf, TU_, mx, TU_, bi, TU_, E_, /*bias*/1);
      } else {
        const __bf16* Kx = krest_bf + (size_t)(l - 1) * (E_ + U_) * TU_;   // rows [0,E)
        const __bf16* Kh = Kx + (size_t)E_ * TU_;                          // rows [E,E+U)
        k_gemm<__bf16><<<grid_gru, blk, 0, stream>>>(
            xt, E_, Kx, TU_, mx, TU_, bi, TU_, E_, /*bias*/1);
        k_gemm<__bf16><<<grid_gru, blk, 0, stream>>>(
            Hcat + (size_t)(l - 1) * U_, L_ * U_, Kh, TU_, mx, TU_, nullptr,
            TU_, U_, /*accumulate*/2);
      }
      // mh = h @ R + br  (h = this layer's previous-timestep state, still in Hcat)
      k_gemm<__bf16><<<grid_gru, blk, 0, stream>>>(
          Hcat + (size_t)l * U_, L_ * U_, rec_bf + (size_t)l * U_ * TU_, TU_,
          mh, TU_, br, TU_, U_, /*bias*/1);
      // gates + state update (writes bf16 h back into Hcat at column l*U)
      k_gru_elem<<<(B_ * U_ + 255) / 256, blk, 0, stream>>>(
          mx, mh, Hf32 + (size_t)l * B_ * U_, Hcat + (size_t)l * U_, L_ * U_);
    }
  }

  // dense head: stream fp32 weights once, convert in-register during staging
  dim3 grid_dense((V_ + BN - 1) / BN, B_ / BM);  // 250 x 2 blocks
  k_gemm<float><<<grid_dense, blk, 0, stream>>>(
      Hcat, L_ * U_, dense_w, V_, out, V_, dense_b, V_, L_ * U_, /*bias*/1);
  k_softmax<<<B_, 256, 0, stream>>>(out, V_);
}
